// Views2Points_79671643340972
// MI455X (gfx1250) — compile-verified, matrix operand order checked by reading, not run on hardware
//
#include <hip/hip_runtime.h>

typedef _Float16 half_t;
typedef __attribute__((ext_vector_type(16))) _Float16 v16h;
typedef __attribute__((ext_vector_type(8)))  float    v8f;
typedef unsigned int u32;
typedef __attribute__((ext_vector_type(4))) u32 v4u;
typedef __attribute__((ext_vector_type(8))) int v8i;
typedef __attribute__((ext_vector_type(4))) int v4i;

#define DD   1024
#define FFD  4096
#define TT   256
#define SSZ  256
#define NNB  16
#define HHN  16
#define HDIM 64
#define TOK  4096   // T*N rows of the activation stream

enum { M_Q = 0, M_K = 1, M_V = 2, M_RES = 3, M_RELU = 4 };

// ---------------------------------------------------------------------------
// fp32 -> fp16 conversion
// ---------------------------------------------------------------------------
__global__ void cvt_f32_to_f16(const float* __restrict__ src,
                               half_t* __restrict__ dst, int n) {
  int i = blockIdx.x * 256 + threadIdx.x;
  if (i < n) dst[i] = (half_t)src[i];
}

// ---------------------------------------------------------------------------
// TDM: async-load a 2D fp16 tile [tile_rows x tile_cols] (row stride in elems)
// from global into LDS. pad_enable inserts 4 DWORDs after every 16 DWORDs
// stored, so a 64-half row lands as 32+8pad+32+8pad = 80-half LDS pitch
// (and a 32-half row as 40-half pitch).
// Descriptor per CDNA5 ISA (D# groups 0/1, type=2).
// ---------------------------------------------------------------------------
__device__ __forceinline__ void tdm_load_tile(u32 lds_addr, const void* gptr,
                                              int tile_rows, int tile_cols,
                                              int row_stride) {
  const unsigned long long ga = (unsigned long long)(uintptr_t)gptr;
  v4u g0;
  g0[0] = 1u;                                  // count=1, user mode
  g0[1] = lds_addr;                            // [63:32] lds_addr
  g0[2] = (u32)ga;                             // global_addr lo
  g0[3] = (u32)((ga >> 32) & 0x01FFFFFFu) | (2u << 30);  // addr hi + type=2
  v8i g1;
  // workgroup_mask=0 | data_size=1(2B) | pad_enable | pad_interval=3 | pad_amount=3
  g1[0] = (1 << 16) | (1 << 20) | (3 << 22) | (3 << 25);
  const u32 td0 = (u32)row_stride;             // tensor_dim0 (never OOB here)
  const u32 td1 = 1u << 20;                    // tensor_dim1: large
  g1[1] = (int)((td0 & 0xFFFFu) << 16);                    // [63:48]=td0 lo16
  g1[2] = (int)((td0 >> 16) | ((td1 & 0xFFFFu) << 16));    // td0 hi / td1 lo
  g1[3] = (int)((td1 >> 16) | ((u32)tile_cols << 16));     // td1 hi / tile_dim0
  g1[4] = (int)(tile_rows & 0xFFFF);                       // tile_dim1, tile_dim2=0
  g1[5] = (int)row_stride;                                 // tensor_dim0_stride lo
  g1[6] = 0;
  g1[7] = 0;
  const v4i z4 = {0, 0, 0, 0};
#if defined(__clang_major__) && (__clang_major__ >= 23)
  const v8i z8 = {0, 0, 0, 0, 0, 0, 0, 0};
  __builtin_amdgcn_tensor_load_to_lds(g0, g1, z4, z4, z8, 0);
#else
  __builtin_amdgcn_tensor_load_to_lds(g0, g1, z4, z4, 0);
#endif
}

// ---------------------------------------------------------------------------
// WMMA GEMM:  C[M=4096, Nout] = A_f16[M,K] * W_f16[Nout,K]^T + bias
//   block tile 128x128, 8 waves (4x2), each wave 32x64.
//   K consumed in chunks of 64: W tile (4x LDS reuse) streamed by TDM into
//   double-buffered LDS; A fragments read directly from global (L2-resident,
//   reused across all column-blocks). 16 WMMA per wave per barrier.
// ---------------------------------------------------------------------------
template<int MODE>
__global__ __launch_bounds__(256)
void gemm_kernel(const half_t* __restrict__ A, const half_t* __restrict__ W,
                 const float* __restrict__ bias, const float* __restrict__ resid,
                 half_t* __restrict__ outh, float* __restrict__ outf,
                 int Nout, int K, float scale)
{
  __shared__ __align__(16) half_t Ws[2][128][80];   // TDM pads 64 -> 80 pitch

  const int tid  = threadIdx.x;
  const int lane = tid & 31;
  const int w    = tid >> 5;
  const int tw   = w & 3;        // wave's 32-row slot within 128
  const int nw   = w >> 2;       // wave's 64-col slot within 128
  const int bm   = blockIdx.y * 128;
  const int bn   = blockIdx.x * 128;
  const int lr   = lane & 15;
  const int lh   = lane >> 4;
  const bool wave0 = (w == 0);

  v8f acc[2][4];
#pragma unroll
  for (int mi = 0; mi < 2; mi++)
#pragma unroll
    for (int ni = 0; ni < 4; ni++) { v8f z = {}; acc[mi][ni] = z; }

  if (wave0) {
    tdm_load_tile((u32)(uintptr_t)&Ws[0][0][0], W + (size_t)bn * K, 128, 64, K);
    __builtin_amdgcn_s_wait_tensorcnt(0);
  }
  __syncthreads();

  // per-wave A base: row = bm + tw*32 + mi*16 + lr, cols follow frag pattern
  const half_t* Arow0 = A + (size_t)(bm + tw * 32 + lr) * K + 8 * lh;
  const half_t* Arow1 = Arow0 + (size_t)16 * K;

  int buf = 0;
  for (int k0 = 0; k0 < K; k0 += 64) {
    const bool has_next = (k0 + 64 < K);
    if (wave0 && has_next) {
      tdm_load_tile((u32)(uintptr_t)&Ws[buf ^ 1][0][0],
                    W + (size_t)bn * K + (k0 + 64), 128, 64, K);
    }
    union { v16h v; uint4 u[2]; } af[2][2], bf[4][2];
#pragma unroll
    for (int kc = 0; kc < 2; kc++) {
      const half_t* a0 = Arow0 + k0 + kc * 32;
      const half_t* a1 = Arow1 + k0 + kc * 32;
      af[0][kc].u[0] = *(const uint4*)a0;
      af[0][kc].u[1] = *(const uint4*)(a0 + 16);
      af[1][kc].u[0] = *(const uint4*)a1;
      af[1][kc].u[1] = *(const uint4*)(a1 + 16);
    }
#pragma unroll
    for (int ni = 0; ni < 4; ni++)
#pragma unroll
      for (int kc = 0; kc < 2; kc++) {
        const half_t* bp = &Ws[buf][nw * 64 + ni * 16 + lr][kc * 40 + 8 * lh];
        bf[ni][kc].u[0] = *(const uint4*)bp;
        bf[ni][kc].u[1] = *(const uint4*)(bp + 16);
      }
#pragma unroll
    for (int kc = 0; kc < 2; kc++)
#pragma unroll
      for (int mi = 0; mi < 2; mi++)
#pragma unroll
        for (int ni = 0; ni < 4; ni++)
          acc[mi][ni] = __builtin_amdgcn_wmma_f32_16x16x32_f16(
              false, af[mi][kc].v, false, bf[ni][kc].v, (short)0,
              acc[mi][ni], false, false);
    if (wave0 && has_next) __builtin_amdgcn_s_wait_tensorcnt(0);
    __syncthreads();
    buf ^= 1;
  }

  // epilogue: C tile layout = VGPR i -> row i + 8*lh, col lr (per 16x16 tile)
#pragma unroll
  for (int mi = 0; mi < 2; mi++) {
#pragma unroll
    for (int i = 0; i < 8; i++) {
      const int gm = bm + tw * 32 + mi * 16 + 8 * lh + i;
#pragma unroll
      for (int ni = 0; ni < 4; ni++) {
        const int gc = bn + nw * 64 + ni * 16 + lr;
        float accv = acc[mi][ni][i] + bias[gc];
        if (MODE == M_Q) {
          const int t = gm >> 4, n = gm & 15, h = gc >> 6, hd = gc & 63;
          outh[(((size_t)(n * HHN + h)) * TT + t) * HDIM + hd] = (half_t)(accv * scale);
        } else if (MODE == M_K) {
          const int sr = gm >> 4, n = gm & 15, h = gc >> 6, hd = gc & 63;
          outh[(((size_t)(n * HHN + h)) * SSZ + sr) * HDIM + hd] = (half_t)accv;
        } else if (MODE == M_V) {  // store V transposed: [nh, hd, s]
          const int sr = gm >> 4, n = gm & 15, h = gc >> 6, hd = gc & 63;
          outh[(((size_t)(n * HHN + h)) * HDIM + hd) * SSZ + sr] = (half_t)accv;
        } else if (MODE == M_RES) {
          outf[(size_t)gm * Nout + gc] = accv + resid[(size_t)gm * Nout + gc];
        } else {  // M_RELU
          outh[(size_t)gm * Nout + gc] = (half_t)(accv > 0.f ? accv : 0.f);
        }
      }
    }
  }
}

// ---------------------------------------------------------------------------
// Attention: one block per (n*H+h, 32-row t tile). QK^T via WMMA into LDS,
// masked softmax in LDS (probs -> fp16), P*V via WMMA, normalize in epilogue.
// ---------------------------------------------------------------------------
template<bool CAUSAL>
__global__ __launch_bounds__(256)
void attn_kernel(const half_t* __restrict__ Q, const half_t* __restrict__ Km,
                 const half_t* __restrict__ Vt, half_t* __restrict__ Out)
{
  __shared__ __align__(16) float  sc[32][260];
  __shared__ __align__(16) half_t pm[32][264];
  __shared__ float red1[32][8];
  __shared__ float red2[32][8];
  __shared__ float linv[32];

  const int nh  = blockIdx.x;          // n*H + h
  const int t0  = blockIdx.y * 32;
  const int tid = threadIdx.x;
  const int lane = tid & 31, w = tid >> 5;
  const int lr = lane & 15, lh = lane >> 4;

  const half_t* Qb = Q  + (size_t)nh * TT  * HDIM;
  const half_t* Kb = Km + (size_t)nh * SSZ * HDIM;
  const half_t* Vb = Vt + (size_t)nh * HDIM * SSZ;

  // ---- phase 1: scores[32][256] = Q(32x64) * K^T ----
  {
    const int tw = w & 1;              // 16 t-rows
    const int sw = w >> 1;             // 64 s-cols
    union { v16h v; uint4 u[2]; } a0, a1;
    const half_t* qp = Qb + (size_t)(t0 + tw * 16 + lr) * HDIM + 8 * lh;
    a0.u[0] = *(const uint4*)qp;         a0.u[1] = *(const uint4*)(qp + 16);
    a1.u[0] = *(const uint4*)(qp + 32);  a1.u[1] = *(const uint4*)(qp + 48);
#pragma unroll
    for (int st = 0; st < 4; st++) {
      const int scol = sw * 64 + st * 16;
      union { v16h v; uint4 u[2]; } b0, b1;
      const half_t* kp = Kb + (size_t)(scol + lr) * HDIM + 8 * lh;
      b0.u[0] = *(const uint4*)kp;         b0.u[1] = *(const uint4*)(kp + 16);
      b1.u[0] = *(const uint4*)(kp + 32);  b1.u[1] = *(const uint4*)(kp + 48);
      v8f c = {};
      c = __builtin_amdgcn_wmma_f32_16x16x32_f16(false, a0.v, false, b0.v,
                                                 (short)0, c, false, false);
      c = __builtin_amdgcn_wmma_f32_16x16x32_f16(false, a1.v, false, b1.v,
                                                 (short)0, c, false, false);
#pragma unroll
      for (int i = 0; i < 8; i++)
        sc[tw * 16 + 8 * lh + i][scol + lr] = c[i];
    }
  }
  __syncthreads();

  // ---- phase 2: softmax (8 threads per row, 32 cols each) ----
  {
    const int row = tid >> 3, seg = tid & 7;
    const int tg  = t0 + row;
    float mx = -3.0e38f;
    for (int j = 0; j < 32; j++) {
      const int c = seg * 32 + j;
      if (!CAUSAL || c <= tg) mx = fmaxf(mx, sc[row][c]);
    }
    red1[row][seg] = mx;
    __syncthreads();
    float rmx = red1[row][0];
    for (int j = 1; j < 8; j++) rmx = fmaxf(rmx, red1[row][j]);
    float sum = 0.f;
    for (int j = 0; j < 32; j++) {
      const int c = seg * 32 + j;
      float e = 0.f;
      if (!CAUSAL || c <= tg) e = __expf(sc[row][c] - rmx);
      pm[row][c] = (half_t)e;
      sum += e;
    }
    red2[row][seg] = sum;
    __syncthreads();
    if (seg == 0) {
      float t = 0.f;
      for (int j = 0; j < 8; j++) t += red2[row][j];
      linv[row] = 1.f / t;
    }
  }
  __syncthreads();

  // ---- phase 3: O(32x64) = P(32x256) * V(256x64), V read transposed ----
  {
    const int tw = w & 1;              // 16 t-rows
    const int hw = w >> 1;             // 16 hd-cols
    v8f c = {};
#pragma unroll
    for (int k0 = 0; k0 < SSZ; k0 += 32) {
      union { v16h v; uint4 u[2]; } a, b;
      const half_t* pp = &pm[tw * 16 + lr][k0 + 8 * lh];
      a.u[0] = *(const uint4*)pp;  a.u[1] = *(const uint4*)(pp + 16);
      const half_t* vp = Vb + (size_t)(hw * 16 + lr) * SSZ + k0 + 8 * lh;
      b.u[0] = *(const uint4*)vp;  b.u[1] = *(const uint4*)(vp + 16);
      c = __builtin_amdgcn_wmma_f32_16x16x32_f16(false, a.v, false, b.v,
                                                 (short)0, c, false, false);
    }
    const int n = nh >> 4;
    const int hcol = (nh & 15) * HDIM + hw * 16 + lr;
#pragma unroll
    for (int i = 0; i < 8; i++) {
      const int trow = tw * 16 + 8 * lh + i;
      const float val = c[i] * linv[trow];
      const int token = (t0 + trow) * NNB + n;
      Out[(size_t)token * DD + hcol] = (half_t)val;
    }
  }
}

// ---------------------------------------------------------------------------
// LayerNorm over D=1024; writes fp32 stream and optional fp16 copy.
// ---------------------------------------------------------------------------
__global__ __launch_bounds__(256)
void ln_kernel(const float* __restrict__ src, const float* __restrict__ g,
               const float* __restrict__ b, float* __restrict__ outf,
               half_t* __restrict__ outh)
{
  __shared__ float s1[256], s2[256];
  const int row = blockIdx.x, tid = threadIdx.x;
  const float* x = src + (size_t)row * DD;
  float v[4]; float sum = 0.f, sq = 0.f;
  for (int j = 0; j < 4; j++) {
    v[j] = x[tid + j * 256]; sum += v[j]; sq += v[j] * v[j];
  }
  s1[tid] = sum; s2[tid] = sq; __syncthreads();
  for (int st = 128; st > 0; st >>= 1) {
    if (tid < st) { s1[tid] += s1[tid + st]; s2[tid] += s2[tid + st]; }
    __syncthreads();
  }
  const float mu   = s1[0] * (1.f / DD);
  const float var  = s2[0] * (1.f / DD) - mu * mu;
  const float rstd = rsqrtf(var + 1e-5f);
  for (int j = 0; j < 4; j++) {
    const int c = tid + j * 256;
    const float y = (v[j] - mu) * rstd * g[c] + b[c];
    outf[(size_t)row * DD + c] = y;
    if (outh) outh[(size_t)row * DD + c] = (half_t)y;
  }
}

// ---------------------------------------------------------------------------
// Host driver
// ---------------------------------------------------------------------------
static inline void cvt(const float* s, half_t* d, int n, hipStream_t st) {
  cvt_f32_to_f16<<<(n + 255) / 256, 256, 0, st>>>(s, d, n);
}

extern "C" void kernel_launch(void* const* d_in, const int* in_sizes, int n_in,
                              void* d_out, int out_size, void* d_ws, size_t ws_size,
                              hipStream_t stream) {
  const float* tgt    = (const float*)d_in[0];
  const float* memory = (const float*)d_in[1];
  const float* sa_w[4] = { (const float*)d_in[2], (const float*)d_in[3],
                           (const float*)d_in[4], (const float*)d_in[5] };
  const float* sa_b[4] = { (const float*)d_in[6], (const float*)d_in[7],
                           (const float*)d_in[8], (const float*)d_in[9] };
  const float* ca_w[4] = { (const float*)d_in[10], (const float*)d_in[11],
                           (const float*)d_in[12], (const float*)d_in[13] };
  const float* ca_b[4] = { (const float*)d_in[14], (const float*)d_in[15],
                           (const float*)d_in[16], (const float*)d_in[17] };
  const float* w1 = (const float*)d_in[18];
  const float* b1 = (const float*)d_in[19];
  const float* w2 = (const float*)d_in[20];
  const float* b2 = (const float*)d_in[21];
  const float* lng[3] = { (const float*)d_in[22], (const float*)d_in[24],
                          (const float*)d_in[26] };
  const float* lnb[3] = { (const float*)d_in[23], (const float*)d_in[25],
                          (const float*)d_in[27] };
  const float* lnf_g = (const float*)d_in[28];
  const float* lnf_b = (const float*)d_in[29];

  char* ws = (char*)d_ws;
  float*  x    = (float*) (ws);                          // 16 MB
  float*  tmp  = (float*) (ws + ((size_t)16 << 20));     // 16 MB
  half_t* xb   = (half_t*)(ws + ((size_t)32 << 20));     //  8 MB
  half_t* memb = (half_t*)(ws + ((size_t)40 << 20));     //  8 MB
  half_t* wb   = (half_t*)(ws + ((size_t)48 << 20));     //  8 MB (weight scratch)
  half_t* qb   = (half_t*)(ws + ((size_t)56 << 20));     //  8 MB
  half_t* kb   = (half_t*)(ws + ((size_t)64 << 20));     //  8 MB
  half_t* vtb  = (half_t*)(ws + ((size_t)72 << 20));     //  8 MB
  half_t* aob  = (half_t*)(ws + ((size_t)80 << 20));     //  8 MB
  half_t* hid  = (half_t*)(ws + ((size_t)88 << 20));     // 32 MB

  const dim3 blk(256);
  const dim3 gD (DD  / 128, TOK / 128);  // Nout=1024 GEMMs: (8, 32)
  const dim3 gFF(FFD / 128, TOK / 128);  // Nout=4096 GEMM: (32, 32)
  const dim3 gAt(NNB * HHN, TT / 32);    // attention
  const float qscale = 0.125f;           // 64^-0.5

  hipMemcpyAsync(x, tgt, (size_t)TOK * DD * sizeof(float),
                 hipMemcpyDeviceToDevice, stream);
  cvt(tgt,    xb,   TOK * DD, stream);
  cvt(memory, memb, TOK * DD, stream);

  for (int l = 0; l < 4; l++) {
    const size_t wo = (size_t)l * DD * DD;
    const size_t bo = (size_t)l * DD;

    // ---- self attention ----
    cvt(sa_w[0] + wo, wb, DD * DD, stream);
    gemm_kernel<M_Q><<<gD, blk, 0, stream>>>(xb, wb, sa_b[0] + bo, nullptr,
                                             qb, nullptr, DD, DD, qscale);
    cvt(sa_w[1] + wo, wb, DD * DD, stream);
    gemm_kernel<M_K><<<gD, blk, 0, stream>>>(xb, wb, sa_b[1] + bo, nullptr,
                                             kb, nullptr, DD, DD, 1.f);
    cvt(sa_w[2] + wo, wb, DD * DD, stream);
    gemm_kernel<M_V><<<gD, blk, 0, stream>>>(xb, wb, sa_b[2] + bo, nullptr,
                                             vtb, nullptr, DD, DD, 1.f);
    attn_kernel<true><<<gAt, blk, 0, stream>>>(qb, kb, vtb, aob);
    cvt(sa_w[3] + wo, wb, DD * DD, stream);
    gemm_kernel<M_RES><<<gD, blk, 0, stream>>>(aob, wb, sa_b[3] + bo, x,
                                               nullptr, tmp, DD, DD, 1.f);
    ln_kernel<<<TOK, blk, 0, stream>>>(tmp, lng[0] + bo, lnb[0] + bo, x, xb);

    // ---- cross attention ----
    cvt(ca_w[0] + wo, wb, DD * DD, stream);
    gemm_kernel<M_Q><<<gD, blk, 0, stream>>>(xb, wb, ca_b[0] + bo, nullptr,
                                             qb, nullptr, DD, DD, qscale);
    cvt(ca_w[1] + wo, wb, DD * DD, stream);
    gemm_kernel<M_K><<<gD, blk, 0, stream>>>(memb, wb, ca_b[1] + bo, nullptr,
                                             kb, nullptr, DD, DD, 1.f);
    cvt(ca_w[2] + wo, wb, DD * DD, stream);
    gemm_kernel<M_V><<<gD, blk, 0, stream>>>(memb, wb, ca_b[2] + bo, nullptr,
                                             vtb, nullptr, DD, DD, 1.f);
    attn_kernel<false><<<gAt, blk, 0, stream>>>(qb, kb, vtb, aob);
    cvt(ca_w[3] + wo, wb, DD * DD, stream);
    gemm_kernel<M_RES><<<gD, blk, 0, stream>>>(aob, wb, ca_b[3] + bo, x,
                                               nullptr, tmp, DD, DD, 1.f);
    ln_kernel<<<TOK, blk, 0, stream>>>(tmp, lng[1] + bo, lnb[1] + bo, x, xb);

    // ---- feed-forward ----
    cvt(w1 + (size_t)l * FFD * DD, wb, FFD * DD, stream);
    gemm_kernel<M_RELU><<<gFF, blk, 0, stream>>>(xb, wb, b1 + (size_t)l * FFD,
                                                 nullptr, hid, nullptr,
                                                 FFD, DD, 1.f);
    cvt(w2 + (size_t)l * DD * FFD, wb, DD * FFD, stream);
    gemm_kernel<M_RES><<<gD, blk, 0, stream>>>(hid, wb, b2 + bo, x,
                                               nullptr, tmp, DD, FFD, 1.f);
    ln_kernel<<<TOK, blk, 0, stream>>>(tmp, lng[2] + bo, lnb[2] + bo, x, xb);
  }

  ln_kernel<<<TOK, blk, 0, stream>>>(x, lnf_g, lnf_b, (float*)d_out, nullptr);
}
